// GroupAttention_6167573037297
// MI455X (gfx1250) — matrix operand order repaired and checked by reading, not compiled, not run on hardware
//
#include <hip/hip_runtime.h>

typedef __attribute__((ext_vector_type(16))) _Float16 v16h;
typedef __attribute__((ext_vector_type(8)))  float    v8f;
typedef __attribute__((ext_vector_type(4)))  float    v4f;

#define S      1024
#define DM     1024
#define NHEAD  16
#define DK     64
#define BS     2
#define NBH    (BS * NHEAD)   // 32

// ---------------------------------------------------------------------------
// Kernel 1: LayerNorm over dm=1024, output as f16 (feeds WMMA A operand).
// One block per row, 256 threads, 4 elements/thread.
// ---------------------------------------------------------------------------
__global__ __launch_bounds__(256) void ln_kernel(
    const float* __restrict__ ctx, const float* __restrict__ gamma,
    const float* __restrict__ beta, _Float16* __restrict__ xh)
{
    const int row = blockIdx.x;            // 0 .. BS*S-1
    const int tid = threadIdx.x;
    const float* x = ctx + (size_t)row * DM;

    float4 xv = *(const float4*)(x + tid * 4);
    float lsum = xv.x + xv.y + xv.z + xv.w;
    float lsq  = xv.x*xv.x + xv.y*xv.y + xv.z*xv.z + xv.w*xv.w;

    __shared__ float ssum[256];
    __shared__ float ssq[256];
    ssum[tid] = lsum; ssq[tid] = lsq;
    __syncthreads();
    for (int off = 128; off > 0; off >>= 1) {
        if (tid < off) { ssum[tid] += ssum[tid+off]; ssq[tid] += ssq[tid+off]; }
        __syncthreads();
    }
    const float mean = ssum[0] * (1.0f / DM);
    const float var  = ssq[0]  * (1.0f / DM) - mean * mean;
    const float rstd = rsqrtf(var + 1e-5f);

    _Float16* o = xh + (size_t)row * DM + tid * 4;
    const int c0 = tid * 4;
    float vals[4] = {xv.x, xv.y, xv.z, xv.w};
    #pragma unroll
    for (int t = 0; t < 4; ++t) {
        float y = (vals[t] - mean) * rstd * gamma[c0 + t] + beta[c0 + t];
        o[t] = (_Float16)y;
    }
}

// ---------------------------------------------------------------------------
// Kernel 2: Q/K projections with V_WMMA_F32_16X16X32_F16.
// One wave per 16-row tile of one (b,h); computes Q[16,64] and K[16,64].
// A layout (16-bit 16x32): lane<16 holds row=lane, K in {0..7,16..23};
//                          lane>=16 holds row=lane-16, K in {8..15,24..31}.
// B layout (32x16 KxN):    lane<16 -> col=lane,   K in {0..7,16..23};
//                          lane>=16 -> col=lane-16, K in {8..15,24..31}.
// D layout: col = lane%16, row = vreg + 8*(lane>=16).
// ---------------------------------------------------------------------------
__global__ __launch_bounds__(256) void qk_proj_kernel(
    const _Float16* __restrict__ xh,
    const float* __restrict__ wq, const float* __restrict__ wk,
    const float* __restrict__ bq, const float* __restrict__ bk,
    float* __restrict__ Q, float* __restrict__ K)
{
    const int lane = threadIdx.x & 31;
    const int wave = threadIdx.x >> 5;
    const int tile = blockIdx.x * 8 + wave;      // 0 .. 2047
    const int rt   = tile & 63;                  // 16-row tile within (b,h)
    const int h    = (tile >> 6) & 15;
    const int b    = tile >> 10;
    const int half = (lane >= 16) ? 1 : 0;
    const int mrow = lane & 15;

    // --- A fragments: x rows (f16), two K chunks (kc = 0, 32) ---
    const size_t xbase = ((size_t)b * S + (size_t)rt * 16 + mrow) * DM + (size_t)h * DK;
    v16h a0, a1;
    #pragma unroll
    for (int t = 0; t < 16; ++t) {
        const int kA = (t & 7) + ((t >> 3) << 4) + (half << 3);
        a0[t] = xh[xbase + kA];
        a1[t] = xh[xbase + 32 + kA];
    }

    #pragma unroll
    for (int m = 0; m < 2; ++m) {
        const float* W    = m ? wk : wq;
        const float* bias = m ? bk : bq;
        float*       Out  = m ? K  : Q;
        #pragma unroll
        for (int nt = 0; nt < 4; ++nt) {
            const int col = nt * 16 + mrow;      // output column / weight row
            v16h b0, b1;
            #pragma unroll
            for (int t = 0; t < 16; ++t) {
                const int kB = (t & 7) + ((t >> 3) << 4) + (half << 3);
                b0[t] = (_Float16)W[col * DK + kB];
                b1[t] = (_Float16)W[col * DK + 32 + kB];
            }
            v8f c = {};
            c = __builtin_amdgcn_wmma_f32_16x16x32_f16(false, a0, false, b0,
                                                       (short)0, c, false, false);
            c = __builtin_amdgcn_wmma_f32_16x16x32_f16(false, a1, false, b1,
                                                       (short)0, c, false, false);
            const float bv = bias[col];
            const size_t obase = (((size_t)(b * NHEAD + h) * S) + (size_t)rt * 16) * DK
                               + nt * 16 + mrow;
            #pragma unroll
            for (int v = 0; v < 8; ++v) {
                const int orow = v + half * 8;
                Out[obase + (size_t)orow * DK] = c[v] + bv;
            }
        }
    }
}

// ---------------------------------------------------------------------------
// Kernel 3: per (b,h) band scores, 2-entry softmax, neibor band values,
// exclusive prefix scan P of L[r] = log(neibor[r,r+1] + 1e-9).
// One 1024-thread block per (b,h).
// ---------------------------------------------------------------------------
__global__ __launch_bounds__(1024) void band_kernel(
    const float* __restrict__ Q, const float* __restrict__ K,
    const int* __restrict__ eos, const float* __restrict__ prior,
    float* __restrict__ P, float* __restrict__ rowU,
    float* __restrict__ rowD, float* __restrict__ degv)
{
    const int bh = blockIdx.x;       // 0..31
    const int b  = bh >> 4;
    const int i  = threadIdx.x;      // 0..1023
    const size_t qbase = ((size_t)bh * S + i) * DK;
    const size_t mbase = (size_t)b * S * S + (size_t)i * S;

    float s_up = 0.f, s_dn = 0.f;
    bool m_up = false, m_dn = false;
    if (i < S - 1) {
        const float* q = Q + qbase;
        const float* k = K + qbase + DK;
        float acc = 0.f;
        #pragma unroll 8
        for (int t = 0; t < DK; ++t) acc += q[t] * k[t];
        s_up = acc * (1.0f / DK);
        m_up = eos[mbase + i + 1] != 0;
    }
    if (i > 0) {
        const float* q = Q + qbase;
        const float* k = K + qbase - DK;
        float acc = 0.f;
        #pragma unroll 8
        for (int t = 0; t < DK; ++t) acc += q[t] * k[t];
        s_dn = acc * (1.0f / DK);
        m_dn = eos[mbase + i - 1] != 0;
    }

    float aU = 0.f, aD = 0.f, dg = 0.f;
    if (m_up && m_dn) {
        const float mx = fmaxf(s_up, s_dn);
        const float eu = __expf(s_up - mx), ed = __expf(s_dn - mx);
        const float inv = 1.0f / (eu + ed);
        aU = eu * inv; aD = ed * inv;
    } else if (m_up) {
        aU = 1.0f;
    } else if (m_dn) {
        aD = 1.0f;
    } else {                           // degenerate row: uniform softmax
        dg = 1.0f / (float)S; aU = dg; aD = dg;
    }

    __shared__ float sD[S];
    __shared__ float sL[S];
    sD[i] = aD;
    __syncthreads();

    float Lv = 0.f;
    if (i < S - 1) {
        const float pU  = aU;          // p[i, i+1]
        const float pDT = sD[i + 1];   // p[i+1, i]
        const float pr  = prior[mbase + i + 1];
        const float nb  = pr + (1.0f - pr) * sqrtf(pU * pDT + 1e-4f);
        Lv = __logf(nb + 1e-9f);
    }

    // exclusive scan of L -> P (Hillis-Steele inclusive, then shift by one)
    sL[i] = Lv;
    __syncthreads();
    #pragma unroll
    for (int off = 1; off < S; off <<= 1) {
        const float v = (i >= off) ? sL[i - off] : 0.f;
        __syncthreads();
        sL[i] += v;
        __syncthreads();
    }
    const float Pv = (i > 0) ? sL[i - 1] : 0.f;

    const size_t o = (size_t)bh * S + i;
    P[o] = Pv; rowU[o] = aU; rowD[o] = aD; degv[o] = dg;
}

// ---------------------------------------------------------------------------
// Kernel 4: bandwidth-bound elementwise pass producing both outputs.
// float4 per thread. 256 MB of streaming writes > 192 MB L2, so use
// non-temporal stores (TH=NT) to keep prior + per-row vectors L2-resident.
// ---------------------------------------------------------------------------
__global__ __launch_bounds__(256) void big_kernel(
    const float* __restrict__ prior, const float* __restrict__ P,
    const float* __restrict__ rowU, const float* __restrict__ rowD,
    const float* __restrict__ degv,
    float* __restrict__ g_out, float* __restrict__ n_out)
{
    const size_t idx = (size_t)blockIdx.x * 256 + threadIdx.x;
    const int jq = (int)(idx & 255);
    const int i  = (int)((idx >> 8) & 1023);
    const int bh = (int)(idx >> 18);
    const int b  = bh >> 4;
    const int j0 = jq * 4;

    const size_t vbase = (size_t)bh * S;
    const float Pi  = P[vbase + i];
    const float rUi = rowU[vbase + i];
    const float rDi = rowD[vbase + i];
    const float dgi = degv[vbase + i];

    const float4 pr4 = *(const float4*)(prior + (size_t)b * S * S + (size_t)i * S + j0);
    const float prs[4] = {pr4.x, pr4.y, pr4.z, pr4.w};

    // contiguous per-j vectors: let the compiler fold into b128 loads
    const float4 Pj4  = *(const float4*)(P    + vbase + j0);
    const float4 dg4  = *(const float4*)(degv + vbase + j0);
    const float Pjs[4]  = {Pj4.x, Pj4.y, Pj4.z, Pj4.w};
    const float dgs[4]  = {dg4.x, dg4.y, dg4.z, dg4.w};

    float go[4], no[4];
    #pragma unroll
    for (int t = 0; t < 4; ++t) {
        const int j = j0 + t;
        const float pij = (j == i + 1) ? rUi
                        : (j == i - 1) ? rDi : dgi;
        float pji = dgs[t];
        if (j == i + 1) pji = rowD[vbase + j];   // p[j,i] with i = j-1
        if (j == i - 1) pji = rowU[vbase + j];   // p[j,i] with i = j+1
        const float pr = prs[t];
        const float nb = pr + (1.0f - pr) * sqrtf(pij * pji + 1e-4f);
        float g;
        if (j == i) {
            g = nb;                                  // diag of where(diag, neibor, ...)
        } else {
            const float d = Pjs[t] - Pi;
            g = __expf((j > i) ? d : -d) + 1e-4f;    // exp(sum of band logs) + 1e-4
        }
        no[t] = nb; go[t] = g;
    }

    const size_t obase = (size_t)bh * (S * S) + (size_t)i * S + j0;
    v4f gg; gg.x = go[0]; gg.y = go[1]; gg.z = go[2]; gg.w = go[3];
    v4f nn; nn.x = no[0]; nn.y = no[1]; nn.z = no[2]; nn.w = no[3];
    __builtin_nontemporal_store(gg, (v4f*)(g_out + obase));
    __builtin_nontemporal_store(nn, (v4f*)(n_out + obase));
}

// ---------------------------------------------------------------------------
extern "C" void kernel_launch(void* const* d_in, const int* in_sizes, int n_in,
                              void* d_out, int out_size, void* d_ws, size_t ws_size,
                              hipStream_t stream)
{
    const float* context = (const float*)d_in[0];   // [2,1024,1024]
    const int*   eos     = (const int*)  d_in[1];   // [2,1,1024,1024]
    const float* prior   = (const float*)d_in[2];   // [2,1,1024,1024]
    const float* gamma   = (const float*)d_in[3];   // [1024]
    const float* beta    = (const float*)d_in[4];   // [1024]
    const float* wq      = (const float*)d_in[5];   // [64,64]
    const float* bq      = (const float*)d_in[6];   // [64]
    const float* wk      = (const float*)d_in[7];   // [64,64]
    const float* bk      = (const float*)d_in[8];   // [64]

    float* g_out = (float*)d_out;                              // [2,16,1024,1024]
    float* n_out = g_out + (size_t)NBH * S * S;                // [2,16,1024,1024]

    // workspace layout
    char* w = (char*)d_ws;
    _Float16* xh = (_Float16*)w;                               // 4 MB
    float* Qb  = (float*)(w + (size_t)(4)  * 1024 * 1024);     // 8 MB
    float* Kb  = (float*)(w + (size_t)(12) * 1024 * 1024);     // 8 MB
    float* Pb  = (float*)(w + (size_t)(20) * 1024 * 1024);     // 128 KB
    float* rU  = Pb + (size_t)NBH * S;
    float* rD  = rU + (size_t)NBH * S;
    float* dgv = rD + (size_t)NBH * S;

    ln_kernel<<<BS * S, 256, 0, stream>>>(context, gamma, beta, xh);
    qk_proj_kernel<<<(BS * NHEAD * (S / 16)) / 8, 256, 0, stream>>>(
        xh, wq, wk, bq, bk, Qb, Kb);
    band_kernel<<<NBH, 1024, 0, stream>>>(Qb, Kb, eos, prior, Pb, rU, rD, dgv);
    big_kernel<<<(unsigned)((size_t)NBH * S * (S / 4) / 256), 256, 0, stream>>>(
        prior, Pb, rU, rD, dgv, g_out, n_out);
}
